// DIN_86234353369605
// MI455X (gfx1250) — matrix operand order repaired and checked by reading, not compiled, not run on hardware
//
#include <hip/hip_runtime.h>
#include <hip/hip_bf16.h>

typedef __attribute__((ext_vector_type(16))) _Float16 v16h;
typedef __attribute__((ext_vector_type(8)))  _Float16 v8h;
typedef __attribute__((ext_vector_type(4)))  _Float16 v4h;
typedef __attribute__((ext_vector_type(8)))  float    v8f;
typedef __attribute__((ext_vector_type(4)))  float    v4f;

#define B_TOT   2048
#define SEQ     200
#define D_ITEM  128
#define D_CAT   64
#define D_TIME  32
#define D_BEH   32
#define D_IN    448
#define HID     256
#define NKC     14        // 448 / 32 K-chunks
#define NB      4         // batch rows per workgroup
#define NTHREADS 512      // 16 waves; wave i owns N-tile i
#define ROW_TILES 13      // ceil(200 / 16)
#define NQ      (D_IN / 4)   // 112 float4 per row
#define TILE_Q  (16 * NQ)    // 1792 float4 per tile
#define LD_ITER 4            // ceil(TILE_Q / NTHREADS)

struct Smem {
    _Float16 a[16 * D_IN];            // A tile, f16, row-major [16][448]
    float stash_item[16 * D_ITEM];    // f32 copies for the output path
    float stash_cat[16 * D_CAT];
    float score[16];
    float total;
};

__global__ __launch_bounds__(NTHREADS, 1)
void din_attn_kernel(const float* __restrict__ item_emb,
                     const float* __restrict__ cat_emb,
                     const float* __restrict__ seq_items,
                     const float* __restrict__ seq_cats,
                     const float* __restrict__ seq_times,
                     const float* __restrict__ seq_beh,
                     const float* __restrict__ mask,
                     const float* __restrict__ W1,
                     const float* __restrict__ b1,
                     const float* __restrict__ W2,
                     const float* __restrict__ b2,
                     float* __restrict__ out)
{
    __shared__ Smem sm;
    const int tid  = threadIdx.x;
    const int lane = tid & 31;
    const int wv   = tid >> 5;            // wave id == N-tile index (0..15)
    const int nlo  = lane & 15;
    const int khi  = (lane >> 4) << 3;    // B-frag K offset: lanes>=16 hold K+8
    const int akb  = (lane >> 4) << 3;    // A-frag K offset: lanes>=16 hold K+8

    // per-lane column constants
    const int   N   = wv * 16 + nlo;
    const float b1v = b1[N];
    const float w2v = W2[N];
    const float b2v = b2[0];

    // ---- convert this wave's W1 N-tile f32 -> f16 WMMA B fragments, resident in VGPRs ----
    // B 32x16 layout: lane (0..15) holds col N=lane, halves 0..7 -> K 0..7, 8..15 -> K 16..23; lanes 16..31 K offset +8.
    v16h breg[NKC];
    {
        const float* wcol = W1 + N;       // W1 is [448][256] row-major
        #pragma unroll
        for (int kc = 0; kc < NKC; ++kc) {
            v16h t;
            #pragma unroll
            for (int h = 0; h < 16; ++h)
                t[h] = (_Float16)wcol[(size_t)(kc * 32 + khi + (h & 7) + ((h >> 3) << 4)) * HID];
            breg[kc] = t;
        }
    }

    // A 16x32 layout: lane (0..15) row M=lane, halves 0..7 -> K 0..7, halves 8..15 -> K 16..23;
    // lanes 16..31 same rows with K offset +8.
    const _Float16* abase = sm.a + (size_t)nlo * D_IN + akb;

    // registers holding the prefetched (next) tile's feature data
    v4f vals[LD_ITER];

    auto load_tile = [&](int bn, int rtn) {
        #pragma unroll
        for (int j = 0; j < LD_ITER; ++j) {
            int i4 = tid + j * NTHREADS;
            v4f v = {0.f, 0.f, 0.f, 0.f};
            if (i4 < TILE_Q) {
                int row = i4 / NQ;
                int k   = (i4 - row * NQ) * 4;
                int s   = rtn * 16 + row;
                if (s < SEQ) {
                    size_t sb = (size_t)bn * SEQ + s;
                    if      (k < 128) v = *(const v4f*)(item_emb + (size_t)bn * D_ITEM + k);
                    else if (k < 256) v = __builtin_nontemporal_load(
                                              (const v4f*)(seq_items + sb * D_ITEM + (k - 128)));
                    else if (k < 320) v = *(const v4f*)(cat_emb + (size_t)bn * D_CAT + (k - 256));
                    else if (k < 384) v = __builtin_nontemporal_load(
                                              (const v4f*)(seq_cats + sb * D_CAT + (k - 320)));
                    else if (k < 416) v = __builtin_nontemporal_load(
                                              (const v4f*)(seq_times + sb * D_TIME + (k - 384)));
                    else              v = __builtin_nontemporal_load(
                                              (const v4f*)(seq_beh + sb * D_BEH + (k - 416)));
                }
            }
            vals[j] = v;
        }
    };

    auto store_tile = [&]() {
        #pragma unroll
        for (int j = 0; j < LD_ITER; ++j) {
            int i4 = tid + j * NTHREADS;
            if (i4 < TILE_Q) {
                int row = i4 / NQ;
                int k   = (i4 - row * NQ) * 4;
                v4f v   = vals[j];
                if      (k >= 128 && k < 256) *(v4f*)(sm.stash_item + row * D_ITEM + (k - 128)) = v;
                else if (k >= 320 && k < 384) *(v4f*)(sm.stash_cat  + row * D_CAT  + (k - 320)) = v;
                v4h h;
                h[0] = (_Float16)v[0]; h[1] = (_Float16)v[1];
                h[2] = (_Float16)v[2]; h[3] = (_Float16)v[3];
                *(v4h*)(sm.a + row * D_IN + k) = h;
            }
        }
    };

    // prefetch the first tile
    load_tile(blockIdx.x * NB, 0);

    for (int bb = 0; bb < NB; ++bb) {
        const int b = blockIdx.x * NB + bb;
        __syncthreads();                   // protect sm reuse from previous batch row
        if (tid == 0) sm.total = 0.f;
        float accum = 0.f;

        for (int rt = 0; rt < ROW_TILES; ++rt) {
            __syncthreads();
            // ---- commit prefetched tile: f16 A + f32 stash ----
            store_tile();
            if (tid < 16) sm.score[tid] = 0.f;
            __syncthreads();

            // ---- prefetch next tile's global data; consumed after next barrier,
            //      so HBM latency is hidden behind the WMMA phase ----
            if      (rt + 1 < ROW_TILES) load_tile(b, rt + 1);
            else if (bb + 1 < NB)        load_tile(b + 1, 0);

            // ---- WMMA: h[16 rows, this wave's 16 cols], f32 accumulate over K=448 ----
            // A fragments double-buffered in registers to overlap ds_load with wmma.
            v8f acc = {};
            v8h lo = *(const v8h*)(abase);
            v8h hi = *(const v8h*)(abase + 16);
            #pragma unroll
            for (int kc = 0; kc < NKC; ++kc) {
                v16h af = __builtin_shufflevector(lo, hi,
                          0,1,2,3,4,5,6,7,8,9,10,11,12,13,14,15);
                if (kc + 1 < NKC) {
                    lo = *(const v8h*)(abase + (kc + 1) * 32);
                    hi = *(const v8h*)(abase + (kc + 1) * 32 + 16);
                }
                acc = __builtin_amdgcn_wmma_f32_16x16x32_f16(
                          false, af, false, breg[kc], (short)0, acc, false, false);
            }

            // ---- silu + W2 weighting; reduce over the 16 columns of this N-tile ----
            float part[8];
            #pragma unroll
            for (int r = 0; r < 8; ++r) {
                float x  = acc[r] + b1v;
                float sl = x / (1.f + __expf(-x));
                part[r]  = sl * w2v;
            }
            #pragma unroll
            for (int m = 1; m < 16; m <<= 1) {
                #pragma unroll
                for (int r = 0; r < 8; ++r)
                    part[r] += __shfl_xor(part[r], m, 32);
            }
            if (nlo == 0) {
                // C/D layout: VGPR r -> row M=r (lanes 0-15) or M=r+8 (lanes 16-31)
                int rb = (lane >> 4) << 3;
                #pragma unroll
                for (int r = 0; r < 8; ++r)
                    atomicAdd(&sm.score[rb + r], part[r]);
            }
            __syncthreads();

            // ---- sigmoid, mask, accumulate unnormalized total ----
            if (tid < 16) {
                int s = rt * 16 + tid;
                float sc = 0.f;
                if (s < SEQ) {
                    float pre = sm.score[tid] + b2v;
                    sc = mask[(size_t)b * SEQ + s] / (1.f + __expf(-pre));
                }
                sm.score[tid] = sc;
                atomicAdd(&sm.total, sc);
            }
            __syncthreads();

            // ---- fused weighted sum: thread t owns output dim t (f32 path) ----
            if (tid < 192) {
                const float* src; int stride;
                if (tid < 128) { src = sm.stash_item + tid;         stride = D_ITEM; }
                else           { src = sm.stash_cat + (tid - 128);  stride = D_CAT;  }
                float a = accum;
                #pragma unroll
                for (int r = 0; r < 16; ++r) a += sm.score[r] * src[r * stride];
                accum = a;
            }
        }

        __syncthreads();
        if (tid < 192)
            out[(size_t)b * 192 + tid] = accum / (sm.total + 1e-8f);
    }
}

extern "C" void kernel_launch(void* const* d_in, const int* in_sizes, int n_in,
                              void* d_out, int out_size, void* d_ws, size_t ws_size,
                              hipStream_t stream) {
    (void)in_sizes; (void)n_in; (void)out_size; (void)d_ws; (void)ws_size;
    const float* item_emb  = (const float*)d_in[0];
    const float* cat_emb   = (const float*)d_in[1];
    const float* seq_items = (const float*)d_in[2];
    const float* seq_cats  = (const float*)d_in[3];
    const float* seq_times = (const float*)d_in[4];
    const float* seq_beh   = (const float*)d_in[5];
    const float* mask      = (const float*)d_in[6];
    const float* W1        = (const float*)d_in[7];
    const float* b1        = (const float*)d_in[8];
    const float* W2        = (const float*)d_in[9];
    const float* b2        = (const float*)d_in[10];
    float* out = (float*)d_out;

    din_attn_kernel<<<dim3(B_TOT / NB), dim3(NTHREADS), 0, stream>>>(
        item_emb, cat_emb, seq_items, seq_cats, seq_times, seq_beh,
        mask, W1, b1, W2, b2, out);
}
